// RelativeMultiHeadAttentionDecoderLayer_56324201120415
// MI455X (gfx1250) — matrix operand order repaired and checked by reading, not compiled, not run on hardware
//
#include <hip/hip_runtime.h>
#include <hip/hip_bf16.h>

// Transformer-XL decoder layer for MI455X (gfx1250), wave32 + WMMA bf16.
// QLEN=1024 MLEN=1024 KLEN=2048 BSZ=4 DMODEL=1024 NHEAD=16 DHEAD=64 DFF=4096
//
// v3: fix async-builtin pointer types (int4 addrspace(1)/(3) per hipcc
// diagnostic). GEMMs consume pre-converted bf16 operands (weights transposed
// into K-pair-packed fragment layout), double-buffered LDS staging via
// GLOBAL_LOAD_ASYNC_TO_LDS_B128, 128x128 block tile / 32x64 wave tile.

#define QLEN 1024
#define MLEN 1024
#define KLENN 2048
#define BSZ 4
#define DMODEL 1024
#define NHEAD 16
#define DHEAD 64
#define DFF 4096

typedef __attribute__((ext_vector_type(16))) __bf16 v16bf;
typedef __attribute__((ext_vector_type(8))) float v8f;

union BF16Frag {
    v16bf v;
    unsigned int w[8];
};

__device__ __forceinline__ unsigned short f2bf(float f) {
    unsigned int u = __float_as_uint(f);
    u += 0x7FFFu + ((u >> 16) & 1u);   // round-to-nearest-even
    return (unsigned short)(u >> 16);
}
__device__ __forceinline__ unsigned int pk2bf(float a, float b) {
    return (unsigned int)f2bf(a) | ((unsigned int)f2bf(b) << 16);
}

// ---- CDNA5 async global->LDS copy (compile-safe guards) --------------------
#if defined(__AMDGCN__) && __has_builtin(__builtin_amdgcn_global_load_async_to_lds_b128)
#define HAVE_ASYNC 1
#else
#define HAVE_ASYNC 0
#endif

typedef int v4i_cdna __attribute__((vector_size(16)));
#define ASYNC_SRC(p) ((__attribute__((address_space(1))) v4i_cdna*)(p))
#define ASYNC_DST(p) ((__attribute__((address_space(3))) v4i_cdna*)(p))

__device__ __forceinline__ void wait_async0() {
#if HAVE_ASYNC
#if __has_builtin(__builtin_amdgcn_s_wait_asynccnt)
    __builtin_amdgcn_s_wait_asynccnt(0);
#else
    asm volatile("s_wait_asynccnt 0" ::: "memory");
#endif
#endif
}

// ---------------------------------------------------------------------------
// GEMM: C[M,N] = A[M,K] @ B[K,N], A = bf16 row-major, B pre-transposed packed:
//   Bt[n][k2] : u32 = bf16 pair (B[2*k2][n], B[2*k2+1][n])
// flags: 1=+bias[N], 2=relu, 4=+resid fp32 (ldc layout), 8=write bf16 to Cbf
// REQUIRES: M%128==0, N%128==0, K%32==0.
// ---------------------------------------------------------------------------
#define GBM 128
#define GBN 128
#define GBK 32

__device__ __forceinline__ void stage_tiles(
        const unsigned short* __restrict__ Abf, const unsigned int* __restrict__ Bt,
        unsigned short* As, unsigned int* Bs,
        int row0, int col0, int k0, int lda, int K2, int tid) {
#pragma unroll
    for (int q = 0; q < 2; ++q) {
        int ca = tid + q * 256;
        int r = ca >> 2;
        int oa = (ca & 3) * 8;                       // 8 bf16 = 16B
        int cb = (ca & 3) * 4;                       // 4 u32  = 16B
#if HAVE_ASYNC
        __builtin_amdgcn_global_load_async_to_lds_b128(
            ASYNC_SRC(Abf + (size_t)(row0 + r) * lda + k0 + oa),
            ASYNC_DST(As + r * GBK + oa), 0, 0);
        __builtin_amdgcn_global_load_async_to_lds_b128(
            ASYNC_SRC(Bt + (size_t)(col0 + r) * K2 + (k0 >> 1) + cb),
            ASYNC_DST(Bs + r * (GBK / 2) + cb), 0, 0);
#else
        *(uint4*)(As + r * GBK + oa) =
            *(const uint4*)(Abf + (size_t)(row0 + r) * lda + k0 + oa);
        *(uint4*)(Bs + r * (GBK / 2) + cb) =
            *(const uint4*)(Bt + (size_t)(col0 + r) * K2 + (k0 >> 1) + cb);
#endif
    }
}

__global__ __launch_bounds__(256)
void gemm_bf16_wmma(const unsigned short* __restrict__ Abf,
                    const unsigned int* __restrict__ Bt,
                    float* __restrict__ C, unsigned short* __restrict__ Cbf,
                    const float* __restrict__ bias, const float* __restrict__ resid,
                    int K, int lda, int ldc, int flags) {
    __shared__ __align__(16) unsigned short As[2][GBM * GBK];       // 2 x 8 KB
    __shared__ __align__(16) unsigned int  Bs[2][GBN * (GBK / 2)];  // 2 x 8 KB

    const int tid  = threadIdx.x;
    const int wave = tid >> 5;
    const int lane = tid & 31;
    const int wm   = wave >> 1;       // 0..3 : 32-row strip
    const int wn   = wave & 1;        // 0..1 : 64-col strip
    const int m16  = lane & 15;
    const int hs   = lane >> 4;
    const int row0 = blockIdx.y * GBM;
    const int col0 = blockIdx.x * GBN;
    const int K2   = K >> 1;

    v8f acc[2][4];
    const v8f vzero = {0.f,0.f,0.f,0.f,0.f,0.f,0.f,0.f};
#pragma unroll
    for (int rt = 0; rt < 2; ++rt)
#pragma unroll
        for (int ct = 0; ct < 4; ++ct) acc[rt][ct] = vzero;

    stage_tiles(Abf, Bt, As[0], Bs[0], row0, col0, 0, lda, K2, tid);

    for (int k0 = 0; k0 < K; k0 += GBK) {
        const int cur = (k0 / GBK) & 1;
        wait_async0();
        __syncthreads();
        if (k0 + GBK < K)
            stage_tiles(Abf, Bt, As[cur ^ 1], Bs[cur ^ 1],
                        row0, col0, k0 + GBK, lda, K2, tid);

        BF16Frag af[2], bfr[4];
#pragma unroll
        for (int rt = 0; rt < 2; ++rt) {
            const unsigned int* ap =
                (const unsigned int*)&As[cur][(wm * 32 + rt * 16 + m16) * GBK];
#pragma unroll
            for (int p = 0; p < 8; ++p) {
                int kk = (hs << 4) + (p << 1);                         // K = 16*hs + 2p
                af[rt].w[p] = ap[kk >> 1];
            }
        }
#pragma unroll
        for (int ct = 0; ct < 4; ++ct) {
            const unsigned int* bp =
                &Bs[cur][(wn * 64 + ct * 16 + m16) * (GBK / 2)];
#pragma unroll
            for (int v = 0; v < 8; ++v)
                bfr[ct].w[v] = bp[(hs << 3) + v];                      // K = 16*hs + 2v
        }
#pragma unroll
        for (int rt = 0; rt < 2; ++rt)
#pragma unroll
            for (int ct = 0; ct < 4; ++ct)
                acc[rt][ct] = __builtin_amdgcn_wmma_f32_16x16x32_bf16(
                    false, af[rt].v, false, bfr[ct].v,
                    (short)0, acc[rt][ct], false, false);
        __syncthreads();
    }

    // Epilogue (C layout: lane holds (row = v + 8*hs, col = m16))
#pragma unroll
    for (int rt = 0; rt < 2; ++rt) {
#pragma unroll
        for (int ct = 0; ct < 4; ++ct) {
            int col = col0 + wn * 64 + ct * 16 + m16;
#pragma unroll
            for (int v = 0; v < 8; ++v) {
                int row = row0 + wm * 32 + rt * 16 + v + (hs << 3);
                float val = acc[rt][ct][v];
                if (flags & 1) val += bias[col];
                if (flags & 4) val += resid[(size_t)row * ldc + col];
                if (flags & 2) val = fmaxf(val, 0.0f);
                if (flags & 8) Cbf[(size_t)row * ldc + col] = f2bf(val);
                else           C[(size_t)row * ldc + col] = val;
            }
        }
    }
}

// ---------------------------------------------------------------------------
// fp32 -> bf16 row-major convert (paired u32 stores). n must be even.
// ---------------------------------------------------------------------------
__global__ __launch_bounds__(256)
void convert_bf16(const float* __restrict__ X, unsigned int* __restrict__ Y, size_t npair) {
    size_t i = (size_t)blockIdx.x * 256 + threadIdx.x;
    if (i >= npair) return;
    Y[i] = pk2bf(X[2 * i], X[2 * i + 1]);
}

// ---------------------------------------------------------------------------
// Weight transpose+pack: W fp32 [K][N] -> Wt u32 [N][K/2] (bf16 k-pairs).
// ---------------------------------------------------------------------------
__global__ __launch_bounds__(256)
void transpose_pack(const float* __restrict__ W, unsigned int* __restrict__ Wt,
                    int K, int N) {
    size_t idx = (size_t)blockIdx.x * 256 + threadIdx.x;
    size_t total = (size_t)N * (K >> 1);
    if (idx >= total) return;
    int n  = (int)(idx % N);          // n fast -> coalesced reads
    int k2 = (int)(idx / N);
    Wt[(size_t)n * (K >> 1) + k2] =
        pk2bf(W[(size_t)(2 * k2) * N + n], W[(size_t)(2 * k2 + 1) * N + n]);
}

// ---------------------------------------------------------------------------
// Pack attention operands from w_heads/r_proj (fp32) into WMMA layouts:
//   Qc/Qp u32 [B*N][QLEN][32]  (q + bias_q / q + bias_k, d-pairs)
//   Kt    u32 [B*N][32][KLEN]  (K^T, d-pairs)
//   Vp    u16 [B*N][KLEN/2][64][2] (V, kpos-pairs)
//   Rt    u32 [N][32][KLEN]    (R^T, d-pairs)
// thread per (j, b, n, d2): total KLEN*BSZ*NHEAD*32 = 4194304
// ---------------------------------------------------------------------------
__global__ __launch_bounds__(256)
void pack_qkvr(const float* __restrict__ wh, const float* __restrict__ r_proj,
               const float* __restrict__ bias_q, const float* __restrict__ bias_k,
               unsigned int* __restrict__ Qc, unsigned int* __restrict__ Qp,
               unsigned int* __restrict__ Kt, unsigned short* __restrict__ Vp,
               unsigned int* __restrict__ Rt) {
    size_t idx = (size_t)blockIdx.x * 256 + threadIdx.x;
    int d2 = (int)(idx & 31);
    size_t t = idx >> 5;
    int n = (int)(t & 15); t >>= 4;
    int b = (int)(t & 3);  t >>= 2;
    int j = (int)t;                               // 0..KLEN-1
    int col = n * DHEAD + 2 * d2;
    size_t whrow = (size_t)(j * BSZ + b) * (3 * DMODEL);
    float q0 = wh[whrow + col],              q1 = wh[whrow + col + 1];
    float k0 = wh[whrow + DMODEL + col],     k1 = wh[whrow + DMODEL + col + 1];
    float v0 = wh[whrow + 2 * DMODEL + col], v1 = wh[whrow + 2 * DMODEL + col + 1];
    int bn = b * NHEAD + n;

    if (j >= MLEN) {
        int i = j - MLEN;
        size_t qi = ((size_t)bn * QLEN + i) * 32 + d2;
        Qc[qi] = pk2bf(q0 + bias_q[col], q1 + bias_q[col + 1]);
        Qp[qi] = pk2bf(q0 + bias_k[col], q1 + bias_k[col + 1]);
    }
    Kt[((size_t)bn * 32 + d2) * KLENN + j] = pk2bf(k0, k1);
    size_t vbase = (((size_t)bn * (KLENN / 2) + (j >> 1)) * DHEAD + 2 * d2) * 2 + (j & 1);
    Vp[vbase]     = f2bf(v0);
    Vp[vbase + 2] = f2bf(v1);
    if (b == 0) {
        Rt[((size_t)n * 32 + d2) * KLENN + j] =
            pk2bf(r_proj[(size_t)j * DMODEL + col], r_proj[(size_t)j * DMODEL + col + 1]);
    }
}

// ---------------------------------------------------------------------------
// Flash-style relative attention (bf16 output for the W_o GEMM).
// Block 128 thr = 4 waves, wave owns 16 q rows; grid (QLEN/64, NHEAD, BSZ).
// rel_shift: BD[i,j] = BDraw[i, j + QLEN-1-i]  (valid for all unmasked j).
// ---------------------------------------------------------------------------
__global__ __launch_bounds__(128)
void attn_rel_wmma(const unsigned int* __restrict__ Qc,
                   const unsigned int* __restrict__ Qp,
                   const unsigned int* __restrict__ Kt,
                   const unsigned int* __restrict__ Vp,
                   const unsigned int* __restrict__ Rt,
                   unsigned short* __restrict__ avbf) {
    __shared__ __align__(16) float bandLDS[4][16][48];
    __shared__ __align__(16) unsigned short pLDS[4][16][32];

    const int wave = threadIdx.x >> 5;
    const int lane = threadIdx.x & 31;
    const int m16 = lane & 15;
    const int hs  = lane >> 4;
    const int qblk = blockIdx.x;
    const int head = blockIdx.y;
    const int batch = blockIdx.z;
    const int bn = batch * NHEAD + head;
    const int i0 = qblk * 64 + wave * 16;
    const float scale = 0.125f;               // 1/sqrt(DHEAD)

    BF16Frag aq[2], ap[2];
    {
        const unsigned int* qc = Qc + ((size_t)bn * QLEN + i0 + m16) * 32;
        const unsigned int* qp = Qp + ((size_t)bn * QLEN + i0 + m16) * 32;
#pragma unroll
        for (int c = 0; c < 2; ++c)
#pragma unroll
            for (int p = 0; p < 8; ++p) {
                int kk = 32 * c + (hs << 4) + (p << 1);
                aq[c].w[p] = qc[kk >> 1];
                ap[c].w[p] = qp[kk >> 1];
            }
    }

    const v8f vzero = {0.f,0.f,0.f,0.f,0.f,0.f,0.f,0.f};
    v8f oacc[4];
#pragma unroll
    for (int dt = 0; dt < 4; ++dt) oacc[dt] = vzero;
    float mrow[8], lrow[8];
#pragma unroll
    for (int v = 0; v < 8; ++v) { mrow[v] = -1e30f; lrow[v] = 0.0f; }

    const unsigned int* KtB = Kt + (size_t)bn * 32 * KLENN;
    const unsigned int* VpB = Vp + (size_t)bn * (KLENN / 2) * DHEAD;
    const unsigned int* RtB = Rt + (size_t)head * 32 * KLENN;

    const int jmaxB = qblk * 64 + 63 + MLEN;       // uniform trip count per block
    const int niter = (jmaxB + 32) >> 5;

    for (int it = 0; it < niter; ++it) {
        const int j0 = it << 5;

        v8f ac[2];
#pragma unroll
        for (int t = 0; t < 2; ++t) {
            v8f a = vzero;
#pragma unroll
            for (int c = 0; c < 2; ++c) {
                BF16Frag bk;
#pragma unroll
                for (int v = 0; v < 8; ++v) {
                    int d2 = (c << 4) + (hs << 3) + v;
                    bk.w[v] = KtB[(size_t)d2 * KLENN + j0 + 16 * t + m16];
                }
                a = __builtin_amdgcn_wmma_f32_16x16x32_bf16(
                    false, aq[c].v, false, bk.v, (short)0, a, false, false);
            }
            ac[t] = a;
        }

        const int jr0 = j0 + (QLEN - 1) - i0 - 15;
#pragma unroll
        for (int tb = 0; tb < 3; ++tb) {
            v8f bd = vzero;
#pragma unroll
            for (int c = 0; c < 2; ++c) {
                BF16Frag br;
#pragma unroll
                for (int v = 0; v < 8; ++v) {
                    int d2 = (c << 4) + (hs << 3) + v;
                    int cjr = jr0 + 16 * tb + m16;
                    cjr = (cjr < 0) ? 0 : ((cjr > KLENN - 1) ? KLENN - 1 : cjr);
                    br.w[v] = RtB[(size_t)d2 * KLENN + cjr];
                }
                bd = __builtin_amdgcn_wmma_f32_16x16x32_bf16(
                    false, ap[c].v, false, br.v, (short)0, bd, false, false);
            }
#pragma unroll
            for (int v = 0; v < 8; ++v)
                bandLDS[wave][v + (hs << 3)][16 * tb + m16] = bd[v];
        }
        __syncthreads();

        float sreg[2][8];
        float rmax[8];
#pragma unroll
        for (int v = 0; v < 8; ++v) rmax[v] = -1e30f;
#pragma unroll
        for (int t = 0; t < 2; ++t)
#pragma unroll
            for (int v = 0; v < 8; ++v) {
                int row = v + (hs << 3);
                int i = i0 + row;
                int j = j0 + 16 * t + m16;
                float sv = -1e30f;
                if (j <= i + MLEN) {
                    int cidx = 16 * t + m16 + 15 - row;
                    sv = (ac[t][v] + bandLDS[wave][row][cidx]) * scale;
                }
                sreg[t][v] = sv;
                rmax[v] = fmaxf(rmax[v], sv);
            }
#pragma unroll
        for (int v = 0; v < 8; ++v) {
            rmax[v] = fmaxf(rmax[v], __shfl_xor(rmax[v], 1));
            rmax[v] = fmaxf(rmax[v], __shfl_xor(rmax[v], 2));
            rmax[v] = fmaxf(rmax[v], __shfl_xor(rmax[v], 4));
            rmax[v] = fmaxf(rmax[v], __shfl_xor(rmax[v], 8));
        }
        float alpha[8];
#pragma unroll
        for (int v = 0; v < 8; ++v) {
            float mnew = fmaxf(mrow[v], rmax[v]);
            alpha[v] = __expf(mrow[v] - mnew);
            mrow[v] = mnew;
        }
        float rsum[8];
#pragma unroll
        for (int v = 0; v < 8; ++v) rsum[v] = 0.0f;
#pragma unroll
        for (int t = 0; t < 2; ++t)
#pragma unroll
            for (int v = 0; v < 8; ++v) {
                float p = __expf(sreg[t][v] - mrow[v]);
                rsum[v] += p;
                pLDS[wave][v + (hs << 3)][16 * t + m16] = f2bf(p);
            }
#pragma unroll
        for (int v = 0; v < 8; ++v) {
            rsum[v] += __shfl_xor(rsum[v], 1);
            rsum[v] += __shfl_xor(rsum[v], 2);
            rsum[v] += __shfl_xor(rsum[v], 4);
            rsum[v] += __shfl_xor(rsum[v], 8);
            lrow[v] = lrow[v] * alpha[v] + rsum[v];
        }
#pragma unroll
        for (int dt = 0; dt < 4; ++dt)
#pragma unroll
            for (int v = 0; v < 8; ++v) oacc[dt][v] *= alpha[v];
        __syncthreads();

        BF16Frag pf;
        {
            const unsigned int* pr = (const unsigned int*)&pLDS[wave][m16][0];
#pragma unroll
            for (int p = 0; p < 8; ++p) {
                int kk = (hs << 4) + (p << 1);
                pf.w[p] = pr[kk >> 1];
            }
        }
#pragma unroll
        for (int dt = 0; dt < 4; ++dt) {
            BF16Frag vf;
#pragma unroll
            for (int v = 0; v < 8; ++v) {
                int k2 = (j0 >> 1) + (hs << 3) + v;
                vf.w[v] = VpB[(size_t)k2 * DHEAD + 16 * dt + m16];
            }
            oacc[dt] = __builtin_amdgcn_wmma_f32_16x16x32_bf16(
                false, pf.v, false, vf.v, (short)0, oacc[dt], false, false);
        }
    }

#pragma unroll
    for (int dt = 0; dt < 4; ++dt)
#pragma unroll
        for (int v = 0; v < 8; ++v) {
            int row = i0 + v + (hs << 3);
            int d = 16 * dt + m16;
            avbf[((size_t)row * BSZ + batch) * DMODEL + head * DHEAD + d] =
                f2bf(oacc[dt][v] / lrow[v]);
        }
}

// ---------------------------------------------------------------------------
// Row LayerNorm: fp32 out + optional bf16 copy. One 256-thr block per row.
// ---------------------------------------------------------------------------
__global__ __launch_bounds__(256)
void layernorm_rows(const float* __restrict__ X, const float* __restrict__ g,
                    const float* __restrict__ b, float* __restrict__ Y,
                    unsigned int* __restrict__ Ybf, int ncol) {
    __shared__ float red[256];
    const int row = blockIdx.x;
    const float* x = X + (size_t)row * ncol;
    float sum = 0.f, sq = 0.f;
    for (int c = threadIdx.x; c < ncol; c += 256) {
        float v = x[c];
        sum += v; sq += v * v;
    }
    red[threadIdx.x] = sum; __syncthreads();
    for (int off = 128; off > 0; off >>= 1) {
        if (threadIdx.x < off) red[threadIdx.x] += red[threadIdx.x + off];
        __syncthreads();
    }
    float mu = red[0] / ncol;
    __syncthreads();
    red[threadIdx.x] = sq; __syncthreads();
    for (int off = 128; off > 0; off >>= 1) {
        if (threadIdx.x < off) red[threadIdx.x] += red[threadIdx.x + off];
        __syncthreads();
    }
    float var = red[0] / ncol - mu * mu;
    float rstd = rsqrtf(var + 1e-5f);
    for (int c = threadIdx.x * 2; c < ncol; c += 512) {
        float y0 = (x[c] - mu) * rstd * g[c] + b[c];
        float y1 = (x[c + 1] - mu) * rstd * g[c + 1] + b[c + 1];
        Y[(size_t)row * ncol + c]     = y0;
        Y[(size_t)row * ncol + c + 1] = y1;
        if (Ybf) Ybf[((size_t)row * ncol + c) >> 1] = pk2bf(y0, y1);
    }
}

// ---------------------------------------------------------------------------
extern "C" void kernel_launch(void* const* d_in, const int* in_sizes, int n_in,
                              void* d_out, int out_size, void* d_ws, size_t ws_size,
                              hipStream_t stream) {
    const float* word_embed = (const float*)d_in[0];
    const float* pos_embed  = (const float*)d_in[1];
    const float* bias_q     = (const float*)d_in[2];
    const float* bias_k     = (const float*)d_in[3];
    const float* memories   = (const float*)d_in[4];
    const float* W_qkv      = (const float*)d_in[5];
    const float* W_r        = (const float*)d_in[6];
    const float* W_o        = (const float*)d_in[7];
    const float* ln1_g      = (const float*)d_in[8];
    const float* ln1_b      = (const float*)d_in[9];
    const float* W_ff1      = (const float*)d_in[10];
    const float* b_ff1      = (const float*)d_in[11];
    const float* W_ff2      = (const float*)d_in[12];
    const float* b_ff2      = (const float*)d_in[13];
    const float* ln2_g      = (const float*)d_in[14];
    const float* ln2_b      = (const float*)d_in[15];
    float* out = (float*)d_out;
    (void)ws_size; (void)in_sizes; (void)n_in; (void)out_size;

    char* ws = (char*)d_ws;
    size_t off = 0;
    auto alloc = [&](size_t bytes) -> void* {
        void* p = ws + off;
        off += (bytes + 255) & ~(size_t)255;
        return p;
    };
    const int MB = MLEN * BSZ;       // 4096
    const int QB = QLEN * BSZ;       // 4096
    float* w_heads = (float*)alloc((size_t)(KLENN * BSZ) * 3 * DMODEL * 4);
    float* r_proj  = (float*)alloc((size_t)KLENN * DMODEL * 4);
    unsigned short* catbf = (unsigned short*)alloc((size_t)(MB + QB) * DMODEL * 2);
    unsigned short* posbf = (unsigned short*)alloc((size_t)KLENN * DMODEL * 2);
    unsigned int* W_qkvt = (unsigned int*)alloc((size_t)(3 * DMODEL) * (DMODEL / 2) * 4);
    unsigned int* W_rt   = (unsigned int*)alloc((size_t)DMODEL * (DMODEL / 2) * 4);
    unsigned int* W_ot   = (unsigned int*)alloc((size_t)DMODEL * (DMODEL / 2) * 4);
    unsigned int* W_ff1t = (unsigned int*)alloc((size_t)DFF * (DMODEL / 2) * 4);
    unsigned int* W_ff2t = (unsigned int*)alloc((size_t)DMODEL * (DFF / 2) * 4);
    unsigned int* Qc = (unsigned int*)alloc((size_t)BSZ * NHEAD * QLEN * 32 * 4);
    unsigned int* Qp = (unsigned int*)alloc((size_t)BSZ * NHEAD * QLEN * 32 * 4);
    unsigned int* Kt = (unsigned int*)alloc((size_t)BSZ * NHEAD * 32 * KLENN * 4);
    unsigned short* Vp = (unsigned short*)alloc((size_t)BSZ * NHEAD * KLENN * DHEAD * 2);
    unsigned int* Rt = (unsigned int*)alloc((size_t)NHEAD * 32 * KLENN * 4);
    unsigned short* avbf   = (unsigned short*)alloc((size_t)QB * DMODEL * 2);
    float* sum1            = (float*)alloc((size_t)QB * DMODEL * 4);
    float* out1            = (float*)alloc((size_t)QB * DMODEL * 4);
    unsigned int* out1bf   = (unsigned int*)alloc((size_t)QB * (DMODEL / 2) * 4);
    unsigned short* ffhbf  = (unsigned short*)alloc((size_t)QB * DFF * 2);
    float* sum2            = (float*)alloc((size_t)QB * DMODEL * 4);

    // ---- one-time operand conversion --------------------------------------
    {
        size_t np = (size_t)MB * DMODEL / 2;
        convert_bf16<<<(unsigned)((np + 255) / 256), 256, 0, stream>>>(
            memories, (unsigned int*)catbf, np);
        convert_bf16<<<(unsigned)((np + 255) / 256), 256, 0, stream>>>(
            word_embed, (unsigned int*)(catbf + (size_t)MB * DMODEL), np);
        size_t nr = (size_t)KLENN * DMODEL / 2;
        convert_bf16<<<(unsigned)((nr + 255) / 256), 256, 0, stream>>>(
            pos_embed, (unsigned int*)posbf, nr);
    }
    {
        auto tp = [&](const float* W, unsigned int* Wt, int K, int N) {
            size_t tot = (size_t)N * (K >> 1);
            transpose_pack<<<(unsigned)((tot + 255) / 256), 256, 0, stream>>>(W, Wt, K, N);
        };
        tp(W_qkv, W_qkvt, DMODEL, 3 * DMODEL);
        tp(W_r,   W_rt,   DMODEL, DMODEL);
        tp(W_o,   W_ot,   DMODEL, DMODEL);
        tp(W_ff1, W_ff1t, DMODEL, DFF);
        tp(W_ff2, W_ff2t, DFF,    DMODEL);
    }

    // 1) w_heads = cat @ W_qkv   [8192 x 3072]
    gemm_bf16_wmma<<<dim3(3 * DMODEL / GBN, (MB + QB) / GBM), 256, 0, stream>>>(
        catbf, W_qkvt, w_heads, nullptr, nullptr, nullptr,
        DMODEL, DMODEL, 3 * DMODEL, 0);
    // 2) r_proj = pos_embed @ W_r   [2048 x 1024]
    gemm_bf16_wmma<<<dim3(DMODEL / GBN, KLENN / GBM), 256, 0, stream>>>(
        posbf, W_rt, r_proj, nullptr, nullptr, nullptr,
        DMODEL, DMODEL, DMODEL, 0);
    // 3) pack attention operands
    pack_qkvr<<<(unsigned)((size_t)KLENN * BSZ * NHEAD * 32 / 256), 256, 0, stream>>>(
        w_heads, r_proj, bias_q, bias_k, Qc, Qp, Kt, Vp, Rt);
    // 4) relative attention -> avbf [QLEN,BSZ,1024] bf16
    attn_rel_wmma<<<dim3(QLEN / 64, NHEAD, BSZ), 128, 0, stream>>>(
        Qc, Qp, Kt, (const unsigned int*)Vp, Rt, avbf);
    // 5) sum1 = word_embed + attn_vec @ W_o ; out1 = LN1(sum1) (+bf16 copy)
    gemm_bf16_wmma<<<dim3(DMODEL / GBN, QB / GBM), 256, 0, stream>>>(
        avbf, W_ot, sum1, nullptr, nullptr, word_embed,
        DMODEL, DMODEL, DMODEL, 4);
    layernorm_rows<<<QB, 256, 0, stream>>>(sum1, ln1_g, ln1_b, out1, out1bf, DMODEL);
    // 6) ffh = relu(out1 @ W_ff1 + b_ff1)  -> bf16
    gemm_bf16_wmma<<<dim3(DFF / GBN, QB / GBM), 256, 0, stream>>>(
        (unsigned short*)out1bf, W_ff1t, nullptr, ffhbf, b_ff1, nullptr,
        DMODEL, DMODEL, DFF, 1 | 2 | 8);
    // 7) sum2 = out1 + ffh @ W_ff2 + b_ff2 ; out = LN2(sum2)
    gemm_bf16_wmma<<<dim3(DMODEL / GBN, QB / GBM), 256, 0, stream>>>(
        ffhbf, W_ff2t, sum2, nullptr, b_ff2, out1,
        DFF, DFF, DMODEL, 1 | 4);
    layernorm_rows<<<QB, 256, 0, stream>>>(sum2, ln2_g, ln2_b, out, nullptr, DMODEL);
}
